// SPECTRELayer_15685220565163
// MI455X (gfx1250) — matrix operand order repaired and checked by reading, not compiled, not run on hardware
//
#include <hip/hip_runtime.h>

typedef __attribute__((ext_vector_type(16))) __bf16 v16bf;
typedef __attribute__((ext_vector_type(8)))  float  v8f;

#define B_  4
#define L_  2048
#define D_  1024
#define H_  16
#define DH_ 64
#define NF_ 1025
#define K_  1024
typedef unsigned short u16;

static __device__ __forceinline__ u16 f2bf(float f) {
  union { float f; unsigned u; } c; c.f = f;
  unsigned r = c.u + 0x7FFFu + ((c.u >> 16) & 1u);   // round-to-nearest-even
  return (u16)(r >> 16);
}
static __device__ __forceinline__ float gelu_exact(float x) {
  return 0.5f * x * (1.0f + erff(x * 0.70710678118654752f));
}

// ---------------- elementwise f32 -> bf16 ----------------
__global__ __launch_bounds__(256) void k_convert(const float* __restrict__ x,
                                                 u16* __restrict__ y, int n) {
  int i = blockIdx.x * 256 + threadIdx.x;
  if (i < n) y[i] = f2bf(x[i]);
}

// ---------------- W (KxN f32) -> WT (NxK bf16), LDS tiled ----------------
__global__ __launch_bounds__(256) void k_twt(const float* __restrict__ W,
                                             u16* __restrict__ WT, int Kd, int Nd) {
  __shared__ float tile[32][33];
  int k0 = blockIdx.x * 32, n0 = blockIdx.y * 32;
  int tx = threadIdx.x & 31, ty = threadIdx.x >> 5;   // ty: 0..7
  for (int i = ty; i < 32; i += 8)
    tile[i][tx] = W[(size_t)(k0 + i) * Nd + n0 + tx];
  __syncthreads();
  for (int i = ty; i < 32; i += 8)
    WT[(size_t)(n0 + i) * Kd + k0 + tx] = f2bf(tile[tx][i]);
}

// ---------------- xbar[b,d] = mean_l x[b,l,d] ----------------
__global__ __launch_bounds__(256) void k_xbar(const float* __restrict__ x,
                                              float* __restrict__ xbar) {
  int idx = blockIdx.x * 256 + threadIdx.x;            // 0..4095
  int b = idx >> 10, d = idx & 1023;
  const float* p = x + (size_t)b * L_ * D_ + d;
  float s = 0.f;
  for (int l = 0; l < L_; ++l) s += p[(size_t)l * D_];
  xbar[idx] = s * (1.0f / (float)L_);
}

// ---------------- per (b,h): q_mean, LN, gate-MLP, wavelet-MLP ----------------
__global__ __launch_bounds__(256) void k_gates(
    const float* __restrict__ xbar, const float* __restrict__ Wq,
    const float* __restrict__ g_ln_w, const float* __restrict__ g_ln_b,
    const float* __restrict__ g_w1, const float* __restrict__ g_b1,
    const float* __restrict__ g_w2, const float* __restrict__ g_b2,
    const float* __restrict__ mod_bias,
    const float* __restrict__ w_ln_w, const float* __restrict__ w_ln_b,
    const float* __restrict__ w_w1, const float* __restrict__ w_b1,
    const float* __restrict__ w_w2, const float* __restrict__ w_b2,
    float* __restrict__ g_re, float* __restrict__ g_im,
    float* __restrict__ wscale) {
  int b = blockIdx.x >> 4, h = blockIdx.x & 15;
  int t = threadIdx.x;
  __shared__ float qm[DH_], lng[DH_], lnw[DH_], h1[4 * DH_], h2[DH_];
  __shared__ float mu, rstd;

  if (t < DH_) {                       // q_mean = xbar @ Wq (column h*64+t)
    const float* xb = xbar + b * D_;
    int col = h * DH_ + t;
    float s = 0.f;
    for (int k = 0; k < D_; ++k) s += xb[k] * Wq[(size_t)k * D_ + col];
    qm[t] = s;
  }
  __syncthreads();
  if (t == 0) {
    float m = 0.f;
    for (int i = 0; i < DH_; ++i) m += qm[i];
    m *= (1.0f / DH_);
    float v = 0.f;
    for (int i = 0; i < DH_; ++i) { float d = qm[i] - m; v += d * d; }
    v *= (1.0f / DH_);
    mu = m; rstd = rsqrtf(v + 1e-5f);
  }
  __syncthreads();
  if (t < DH_) {
    float xn = (qm[t] - mu) * rstd;
    lng[t] = xn * g_ln_w[t] + g_ln_b[t];
    lnw[t] = xn * w_ln_w[t] + w_ln_b[t];
  }
  __syncthreads();
  {                                    // gate MLP layer 1: 64 -> 256, exact GELU
    float s = g_b1[t];
    for (int d = 0; d < DH_; ++d) s += lng[d] * g_w1[d * (4 * DH_) + t];
    h1[t] = gelu_exact(s);
  }
  if (t < DH_) {                       // wavelet MLP layer 1: 64 -> 64
    float s = w_b1[t];
    for (int d = 0; d < DH_; ++d) s += lnw[d] * w_w1[d * DH_ + t];
    h2[t] = gelu_exact(s);
  }
  __syncthreads();
  int gbase = (b * H_ + h) * NF_;
  for (int f = t; f < NF_; f += 256) { // gate layer 2 + complex modulation
    float sre = g_b2[f], sim = g_b2[f + NF_];
    for (int j = 0; j < 4 * DH_; ++j) {
      float hv = h1[j];
      sre += hv * g_w2[(size_t)j * (2 * NF_) + f];
      sim += hv * g_w2[(size_t)j * (2 * NF_) + f + NF_];
    }
    float mag = sqrtf(sre * sre + sim * sim);
    float sc = fmaxf(mag + mod_bias[f], 0.f) / (mag + 1e-6f);
    g_re[gbase + f] = sc * sre;
    g_im[gbase + f] = sc * sim;
  }
  if (t < DH_) {                       // wgate -> store (1 + sigmoid(.))
    float s = w_b2[t];
    for (int j = 0; j < DH_; ++j) s += h2[j] * w_w2[j * DH_ + t];
    wscale[(b * H_ + h) * DH_ + t] = 1.f + 1.f / (1.f + expf(-s));
  }
}

// ---------------- WMMA bf16 GEMM: C[MxN] = A[MxK] * BT[NxK]^T ----------------
// MODE 0: C row-major MxN.  MODE 1: scatter into v_t laid out (B, D, L).
template <int MODE>
__global__ __launch_bounds__(128) void k_gemm(const u16* __restrict__ A,
                                              const u16* __restrict__ BT,
                                              float* __restrict__ C,
                                              int M, int N) {
  const int lane = threadIdx.x & 31;
  const int w    = threadIdx.x >> 5;
  const int r    = lane & 15;
  const int hi   = lane >> 4;
  const int wm = blockIdx.x * 128 + (w >> 1) * 64;
  const int wn = blockIdx.y * 128 + (w & 1) * 64;

  v8f acc[4][4];
#pragma unroll
  for (int i = 0; i < 4; ++i)
#pragma unroll
    for (int j = 0; j < 4; ++j)
#pragma unroll
      for (int e = 0; e < 8; ++e) acc[i][j][e] = 0.f;

  for (int k = 0; k < K_; k += 32) {
    union { v16bf v; uint4 q[2]; } a[4], b[4];
#pragma unroll
    for (int i = 0; i < 4; ++i) {
      // A 16x32 bf16 fragment: lane<16 rows m, K {k..k+7, k+16..k+23};
      // lane>=16 same rows, K {k+8..k+15, k+24..k+31}
      const u16* p = A + (size_t)(wm + 16 * i + r) * K_ + k + hi * 8;
      a[i].q[0] = *(const uint4*)p;
      a[i].q[1] = *(const uint4*)(p + 16);
    }
#pragma unroll
    for (int j = 0; j < 4; ++j) {
      // B 32x16 fragment from pre-transposed weight (NxK):
      // lane<16 holds col n, K {k..k+7, k+16..k+23} ; lane>=16 K {k+8..k+15, k+24..k+31}
      const u16* p = BT + (size_t)(wn + 16 * j + r) * K_ + k + hi * 8;
      b[j].q[0] = *(const uint4*)p;
      b[j].q[1] = *(const uint4*)(p + 16);
    }
#pragma unroll
    for (int i = 0; i < 4; ++i)
#pragma unroll
      for (int j = 0; j < 4; ++j)
        acc[i][j] = __builtin_amdgcn_wmma_f32_16x16x32_bf16(
            false, a[i].v, false, b[j].v, (short)0, acc[i][j], false, false);
  }

#pragma unroll
  for (int i = 0; i < 4; ++i)
#pragma unroll
    for (int j = 0; j < 4; ++j) {
      int n  = wn + 16 * j + r;
      int mb = wm + 16 * i + hi * 8;
#pragma unroll
      for (int e = 0; e < 8; ++e) {
        int m = mb + e;
        if (MODE == 0) {
          C[(size_t)m * N + n] = acc[i][j][e];
        } else {
          // m = b*L + l ; store v_t[(b*D + n)*L + l]
          C[((size_t)(m >> 11) * D_ + n) * L_ + (m & (L_ - 1))] = acc[i][j][e];
        }
      }
    }
}

// ---------------- per-column 2048-pt FFT -> gate -> IFFT (in place) ----------
struct c32 { float x, y; };

__global__ __launch_bounds__(256) void k_fft(float* __restrict__ vt,
                                             const float* __restrict__ g_re,
                                             const float* __restrict__ g_im) {
  __shared__ c32 bufA[L_];
  __shared__ c32 bufB[L_];
  const int t = threadIdx.x;
  const size_t base = (size_t)blockIdx.x * L_;
  const int chan = blockIdx.x & (D_ - 1);
  const int b    = blockIdx.x >> 10;
  const int gbase = (b * H_ + (chan >> 6)) * NF_;

  for (int i = t; i < L_; i += 256) { bufA[i].x = vt[base + i]; bufA[i].y = 0.f; }
  __syncthreads();

  c32* src = bufA;
  c32* dst = bufB;
  // forward Stockham radix-2, w = exp(-2*pi*i/n)
  {
    int n = L_, s = 1, ls = 0;
    while (n > 1) {
      int m = n >> 1;
      for (int j = t; j < (L_ >> 1); j += 256) {
        int q = j & (s - 1);
        int p = j >> ls;
        float ang = -6.2831853071795864f * (float)p / (float)n;
        float sn, cs; __sincosf(ang, &sn, &cs);
        c32 a = src[q + (p << ls)];
        c32 c = src[q + ((p + m) << ls)];
        c32 sum = { a.x + c.x, a.y + c.y };
        c32 dif = { a.x - c.x, a.y - c.y };
        dst[q + ((2 * p) << ls)]     = sum;
        dst[q + ((2 * p + 1) << ls)] = { dif.x * cs - dif.y * sn,
                                         dif.x * sn + dif.y * cs };
      }
      __syncthreads();
      c32* tmp = src; src = dst; dst = tmp;
      n = m; s <<= 1; ++ls;
    }
  }
  // spectrum gate multiply (Hermitian extension of g for k > L/2)
  for (int k = t; k < L_; k += 256) {
    float gr, gi;
    if (k <= L_ / 2) { gr = g_re[gbase + k];      gi =  g_im[gbase + k]; }
    else             { gr = g_re[gbase + L_ - k]; gi = -g_im[gbase + L_ - k]; }
    c32 X = src[k];
    src[k] = { X.x * gr - X.y * gi, X.x * gi + X.y * gr };
  }
  __syncthreads();
  // inverse: conjugate twiddles, 1/L scale at the end
  {
    int n = L_, s = 1, ls = 0;
    while (n > 1) {
      int m = n >> 1;
      for (int j = t; j < (L_ >> 1); j += 256) {
        int q = j & (s - 1);
        int p = j >> ls;
        float ang = 6.2831853071795864f * (float)p / (float)n;
        float sn, cs; __sincosf(ang, &sn, &cs);
        c32 a = src[q + (p << ls)];
        c32 c = src[q + ((p + m) << ls)];
        c32 sum = { a.x + c.x, a.y + c.y };
        c32 dif = { a.x - c.x, a.y - c.y };
        dst[q + ((2 * p) << ls)]     = sum;
        dst[q + ((2 * p + 1) << ls)] = { dif.x * cs - dif.y * sn,
                                         dif.x * sn + dif.y * cs };
      }
      __syncthreads();
      c32* tmp = src; src = dst; dst = tmp;
      n = m; s <<= 1; ++ls;
    }
  }
  for (int i = t; i < L_; i += 256)
    vt[base + i] = src[i].x * (1.0f / (float)L_);
}

// ---- v_t (B,D,L) * (1+wgate) -> y bf16 (B,L,D), LDS-tiled transpose --------
__global__ __launch_bounds__(256) void k_tys(const float* __restrict__ vt,
                                             const float* __restrict__ wscale,
                                             u16* __restrict__ y) {
  __shared__ float tile[32][33];
  int l0 = blockIdx.x * 32, d0 = blockIdx.y * 32, b = blockIdx.z;
  int tx = threadIdx.x & 31, ty = threadIdx.x >> 5;
  for (int i = ty; i < 32; i += 8)
    tile[i][tx] = vt[((size_t)b * D_ + d0 + i) * L_ + l0 + tx];
  __syncthreads();
  for (int i = ty; i < 32; i += 8) {
    int d = d0 + tx, l = l0 + i;
    y[((size_t)b * L_ + l) * D_ + d] = f2bf(tile[tx][i] * wscale[b * D_ + d]);
  }
}

// ============================ host launcher =================================
extern "C" void kernel_launch(void* const* d_in, const int* in_sizes, int n_in,
                              void* d_out, int out_size, void* d_ws, size_t ws_size,
                              hipStream_t stream) {
  const float* x      = (const float*)d_in[0];
  const float* Wq     = (const float*)d_in[1];
  const float* Wv     = (const float*)d_in[2];
  const float* Wo     = (const float*)d_in[3];
  const float* g_ln_w = (const float*)d_in[4];
  const float* g_ln_b = (const float*)d_in[5];
  const float* g_w1   = (const float*)d_in[6];
  const float* g_b1   = (const float*)d_in[7];
  const float* g_w2   = (const float*)d_in[8];
  const float* g_b2   = (const float*)d_in[9];
  const float* mod_b  = (const float*)d_in[10];
  const float* w_ln_w = (const float*)d_in[11];
  const float* w_ln_b = (const float*)d_in[12];
  const float* w_w1   = (const float*)d_in[13];
  const float* w_b1   = (const float*)d_in[14];
  const float* w_w2   = (const float*)d_in[15];
  const float* w_b2   = (const float*)d_in[16];

  char* ws = (char*)d_ws;
  u16*   xb16 = (u16*)(ws + 0);               // 16,777,216 B
  u16*   WvT  = (u16*)(ws + 16777216);        //  2,097,152 B
  u16*   WoT  = (u16*)(ws + 18874368);        //  2,097,152 B
  float* vt   = (float*)(ws + 20971520);      // 33,554,432 B  (B,D,L)
  u16*   y    = (u16*)(ws + 54525952);        // 16,777,216 B  (B,L,D)
  float* xbar = (float*)(ws + 71303168);      //     16,384 B
  float* gre  = (float*)(ws + 71319552);      //    262,400 B
  float* gim  = (float*)(ws + 71581952);      //    262,400 B
  float* wsc  = (float*)(ws + 71844352);      //     16,384 B

  const int nX = B_ * L_ * D_;
  k_convert<<<(nX + 255) / 256, 256, 0, stream>>>(x, xb16, nX);
  k_twt<<<dim3(K_ / 32, D_ / 32), 256, 0, stream>>>(Wv, WvT, K_, D_);
  k_twt<<<dim3(K_ / 32, D_ / 32), 256, 0, stream>>>(Wo, WoT, K_, D_);
  k_xbar<<<(B_ * D_) / 256, 256, 0, stream>>>(x, xbar);
  k_gates<<<B_ * H_, 256, 0, stream>>>(xbar, Wq, g_ln_w, g_ln_b, g_w1, g_b1,
                                       g_w2, g_b2, mod_b, w_ln_w, w_ln_b,
                                       w_w1, w_b1, w_w2, w_b2, gre, gim, wsc);
  // v = x @ Wv  (scatter into (B,D,L))
  k_gemm<1><<<dim3((B_ * L_) / 128, D_ / 128), 128, 0, stream>>>(
      xb16, WvT, vt, B_ * L_, D_);
  // spectral mixing per column
  k_fft<<<B_ * D_, 256, 0, stream>>>(vt, gre, gim);
  // wavelet gate (exactly v_mix * (1+wgate)) + transpose + bf16
  k_tys<<<dim3(L_ / 32, D_ / 32, B_), 256, 0, stream>>>(vt, wsc, y);
  // out = y @ Wo
  k_gemm<0><<<dim3((B_ * L_) / 128, D_ / 128), 128, 0, stream>>>(
      y, WoT, (float*)d_out, B_ * L_, D_);
}
